// SimpleCrossAttention_75161927680560
// MI455X (gfx1250) — matrix operand order repaired and checked by reading, not compiled, not run on hardware
//
#include <hip/hip_runtime.h>

// ---------------------------------------------------------------------------
// SimpleCrossAttention on MI455X (gfx1250): bf16 WMMA pipeline, async-LDS K
//   Q = qf@Wq^T+bq ; K = kf@Wk^T+bk ; V = kf@Wv^T+bv  (V stored transposed)
//   S = mask(QK^T/16) -> attn region of d_out ; softmax rows in place
//   ctx = S@V (16x128 wave tiles, single attn sweep per 128 cols)
//   attended = ctx@Wo^T + bo
// ---------------------------------------------------------------------------

#define NQ   8192
#define NK   8192
#define DIMD 256

typedef __attribute__((ext_vector_type(4)))  float   v4f;
typedef __attribute__((ext_vector_type(8)))  float   v8f;
typedef __attribute__((ext_vector_type(8)))  __bf16  v8bf;
typedef __attribute__((ext_vector_type(16))) __bf16  v16bf;

// ---- CDNA5 async global->LDS (ASYNCcnt) ------------------------------------
__device__ __forceinline__ void g2l_async_b128(__bf16* lds_dst, const __bf16* g_src) {
#if defined(__AMDGCN__)
  unsigned int off = (unsigned int)(unsigned long long)lds_dst;  // LDS byte offset
  asm volatile("global_load_async_to_lds_b128 %0, %1, off"
               :: "v"(off), "v"(g_src) : "memory");
#else
  *(v8bf*)lds_dst = *(const v8bf*)g_src;
#endif
}

__device__ __forceinline__ void wait_async0() {
#if defined(__AMDGCN__)
  asm volatile("s_wait_asynccnt 0x0" ::: "memory");
#endif
}

// ---- fragment loaders (CDNA5 ISA 7.12.2 layouts, wave32) -------------------

// A fragment (16x32 bf16 row major): lane m=lane&15, kh=lane>>4.
// elems 0..15 <- K=k0+kh*16 .. +15 (contiguous, matches B)
__device__ __forceinline__ v16bf load_a_bf16(const __bf16* p, long ld,
                                             int m0, int k0, int lane) {
  int m  = m0 + (lane & 15);
  int kh = (lane >> 4) * 16;
  const __bf16* r = p + (long)m * ld + k0 + kh;
  v8bf lo = *(const v8bf*)(r);
  v8bf hi = *(const v8bf*)(r + 8);
  return __builtin_shufflevector(lo, hi, 0,1,2,3,4,5,6,7,8,9,10,11,12,13,14,15);
}

// A fragment from fp32 source (convert to bf16 in registers)
__device__ __forceinline__ v16bf load_a_f32(const float* p, long ld,
                                            int m0, int k0, int lane) {
  int m  = m0 + (lane & 15);
  int kh = (lane >> 4) * 16;
  const float* r = p + (long)m * ld + k0 + kh;
  v4f a0 = *(const v4f*)(r);
  v4f a1 = *(const v4f*)(r + 4);
  v4f a2 = *(const v4f*)(r + 8);
  v4f a3 = *(const v4f*)(r + 12);
  v16bf o;
#pragma unroll
  for (int i = 0; i < 4; ++i) {
    o[i]      = (__bf16)a0[i];
    o[4 + i]  = (__bf16)a1[i];
    o[8 + i]  = (__bf16)a2[i];
    o[12 + i] = (__bf16)a3[i];
  }
  return o;
}

// B fragment (32x16) where B = X^T: lane n=lane&15 selects row n0+n of X,
// VGPRs hold K = k0 + (lane>>4)*16 .. +15  -> one contiguous 32B load.
__device__ __forceinline__ v16bf load_bT(const __bf16* x, long ldx,
                                         int n0, int k0, int lane) {
  int n  = n0 + (lane & 15);
  int kh = (lane >> 4) * 16;
  return *(const v16bf*)(x + (long)n * ldx + k0 + kh);
}

#define WMMA_BF16(a, b, c) \
  __builtin_amdgcn_wmma_f32_16x16x32_bf16(false, (a), false, (b), (short)0, (c), false, false)

// ---- fp32 -> bf16 elementwise convert --------------------------------------
__global__ void cvt_f32_bf16(const float* __restrict__ src,
                             __bf16* __restrict__ dst, long n) {
  long i = ((long)blockIdx.x * blockDim.x + threadIdx.x) * 8;
  if (i + 8 <= n) {
    v4f a = *(const v4f*)(src + i);
    v4f b = *(const v4f*)(src + i + 4);
    v8bf o;
#pragma unroll
    for (int j = 0; j < 4; ++j) { o[j] = (__bf16)a[j]; o[4 + j] = (__bf16)b[j]; }
    *(v8bf*)(dst + i) = o;
  }
}

// ---- Y = X @ W^T + bias  (X: rows x 256 bf16, W: 256x256 bf16) -------------
// mode 0: bf16 row-major  mode 1: bf16 transposed (dstT[col*rows+row])
// mode 2: fp32 row-major
__global__ void gemm_xWT_bias(const __bf16* __restrict__ X,
                              const __bf16* __restrict__ W,
                              const float* __restrict__ bias,
                              void* __restrict__ dst, int rows, int mode) {
  int lane = threadIdx.x & 31;
  int wave = threadIdx.x >> 5;
  int m0 = blockIdx.y * 128 + wave * 16;
  int n0 = blockIdx.x * 64;
  v8f acc[4] = {};
  for (int k0 = 0; k0 < DIMD; k0 += 32) {
    v16bf a = load_a_bf16(X, DIMD, m0, k0, lane);
#pragma unroll
    for (int t = 0; t < 4; ++t) {
      v16bf b = load_bT(W, DIMD, n0 + 16 * t, k0, lane);
      acc[t] = WMMA_BF16(a, b, acc[t]);
    }
  }
  int rbase = m0 + ((lane >> 4) << 3);
  int cl = lane & 15;
#pragma unroll
  for (int t = 0; t < 4; ++t) {
    int col = n0 + 16 * t + cl;
    float bv = bias[col];
#pragma unroll
    for (int v = 0; v < 8; ++v) {
      int row = rbase + v;
      float val = acc[t][v] + bv;
      if (mode == 0)      ((__bf16*)dst)[(long)row * DIMD + col] = (__bf16)val;
      else if (mode == 1) ((__bf16*)dst)[(long)col * rows + row] = (__bf16)val;
      else                ((float*)dst)[(long)row * DIMD + col] = val;
    }
  }
}

// ---- S = mask(Q @ K^T * scale), invalid -> -inf ----------------------------
// Block: 128 q-rows x 128 k-cols. K tile staged in LDS via async loads,
// half of K-dim (128) at a time: 128 rows x 136-elem padded stride = 34 KB.
#define KSTR 136   // padded LDS row stride (272 B: distinct banks, 16B aligned)
__global__ void score_kernel(const __bf16* __restrict__ Q,
                             const __bf16* __restrict__ Kb,
                             const int* __restrict__ qbat,
                             const int* __restrict__ kbat,
                             float* __restrict__ S) {
  __shared__ __bf16 smemK[128 * KSTR];
  const int tid  = threadIdx.x;
  const int lane = tid & 31;
  const int wave = tid >> 5;
  const int m0    = blockIdx.y * 128 + wave * 16;
  const int nbase = blockIdx.x * 128;
  v8f acc[8] = {};
  const int nl = lane & 15;
  const int kh = (lane >> 4) * 16;

  for (int half = 0; half < 2; ++half) {
    __syncthreads();   // previous half fully consumed before overwrite
    // stage K[nbase..+127][half*128..+127]: 2048 x 16B chunks, 8 per thread
#pragma unroll
    for (int i = 0; i < 8; ++i) {
      int c   = tid + 256 * i;     // 0..2047
      int row = c >> 4;            // 0..127
      int seg = c & 15;            // 16B segment within the 256B row slice
      g2l_async_b128(&smemK[row * KSTR + seg * 8],
                     Kb + (long)(nbase + row) * DIMD + half * 128 + seg * 8);
    }
    wait_async0();
    __syncthreads();
    for (int kk = 0; kk < 128; kk += 32) {
      int k0 = half * 128 + kk;
      v16bf a = load_a_bf16(Q, DIMD, m0, k0, lane);
#pragma unroll
      for (int t = 0; t < 8; ++t) {
        v16bf b = *(const v16bf*)&smemK[(nl + 16 * t) * KSTR + kk + kh];
        acc[t] = WMMA_BF16(a, b, acc[t]);
      }
    }
  }

  const float scale = 0.0625f;  // 1/sqrt(256)
  const float ninf = -__builtin_inff();
  int rbase = m0 + ((lane >> 4) << 3);
#pragma unroll
  for (int t = 0; t < 8; ++t) {
    int col = nbase + 16 * t + nl;
    int kb = kbat[col];
#pragma unroll
    for (int v = 0; v < 8; ++v) {
      int row = rbase + v;
      float val = (qbat[row] == kb) ? acc[t][v] * scale : ninf;
      S[(long)row * NK + col] = val;
    }
  }
}

// ---- per-row masked softmax, in place (1 block = 1 row, 256 threads) -------
__global__ void softmax_kernel(float* __restrict__ S) {
  __shared__ float red[256];
  int t = threadIdx.x;
  float* r = S + (long)blockIdx.x * NK;
  float v[32];
  float mx = -__builtin_inff();
#pragma unroll
  for (int c = 0; c < 8; ++c) {
    v4f x = *(const v4f*)(r + (c * 256 + t) * 4);
#pragma unroll
    for (int j = 0; j < 4; ++j) {
      v[c * 4 + j] = x[j];
      mx = fmaxf(mx, x[j]);
    }
  }
  red[t] = mx; __syncthreads();
  for (int s = 128; s > 0; s >>= 1) {
    if (t < s) red[t] = fmaxf(red[t], red[t + s]);
    __syncthreads();
  }
  mx = red[0]; __syncthreads();

  bool valid = (mx != -__builtin_inff());
  float sum = 0.f;
#pragma unroll
  for (int i = 0; i < 32; ++i) {
    float e = valid ? __expf(v[i] - mx) : 0.f;   // exp(-inf)=0 for masked cols
    v[i] = e;
    sum += e;
  }
  red[t] = sum; __syncthreads();
  for (int s = 128; s > 0; s >>= 1) {
    if (t < s) red[t] += red[t + s];
    __syncthreads();
  }
  sum = red[0];
  float inv = (valid && sum > 0.f) ? 1.0f / sum : 0.f;  // nan_to_num rows -> 0
#pragma unroll
  for (int c = 0; c < 8; ++c) {
    v4f y;
#pragma unroll
    for (int j = 0; j < 4; ++j) y[j] = v[c * 4 + j] * inv;
    *(v4f*)(r + (c * 256 + t) * 4) = y;
  }
}

// ---- ctx = attn @ V  (attn fp32 cvt->bf16 on the fly; V given transposed) --
// 16x128 wave tile: attn (256MB, > L2) streamed only twice total.
__global__ void ctx_kernel(const float* __restrict__ S,
                           const __bf16* __restrict__ Vt,
                           __bf16* __restrict__ C) {
  int lane = threadIdx.x & 31;
  int wave = threadIdx.x >> 5;
  int m0 = blockIdx.y * 128 + wave * 16;
  int n0 = blockIdx.x * 128;
  v8f acc[8] = {};
  const float* arow = S + (long)(m0 + (lane & 15)) * NK;
  for (int k0 = 0; k0 < NK; k0 += 32) {
    __builtin_prefetch(arow + k0 + 512, 0, 1);     // global_prefetch_b8
    v16bf a = load_a_f32(S, (long)NK, m0, k0, lane);
#pragma unroll
    for (int t = 0; t < 8; ++t) {
      v16bf b = load_bT(Vt, (long)NK, n0 + 16 * t, k0, lane);
      acc[t] = WMMA_BF16(a, b, acc[t]);
    }
  }
  int rbase = m0 + ((lane >> 4) << 3);
  int cl = lane & 15;
#pragma unroll
  for (int t = 0; t < 8; ++t) {
    int col = n0 + 16 * t + cl;
#pragma unroll
    for (int v = 0; v < 8; ++v)
      C[(long)(rbase + v) * DIMD + col] = (__bf16)acc[t][v];
  }
}

// ---------------------------------------------------------------------------
extern "C" void kernel_launch(void* const* d_in, const int* in_sizes, int n_in,
                              void* d_out, int out_size, void* d_ws, size_t ws_size,
                              hipStream_t stream) {
  (void)in_sizes; (void)n_in; (void)out_size; (void)ws_size;
  const float* q_feat = (const float*)d_in[0];
  const float* k_feat = (const float*)d_in[1];
  const int*   qbat   = (const int*)d_in[2];
  const int*   kbat   = (const int*)d_in[3];
  const float* Wq = (const float*)d_in[4];
  const float* bq = (const float*)d_in[5];
  const float* Wk = (const float*)d_in[6];
  const float* bk = (const float*)d_in[7];
  const float* Wv = (const float*)d_in[8];
  const float* bv = (const float*)d_in[9];
  const float* Wo = (const float*)d_in[10];
  const float* bo = (const float*)d_in[11];

  float* out      = (float*)d_out;
  float* attended = out;                        // [NQ, DIMD]
  float* attn     = out + (size_t)NQ * DIMD;    // [NQ, NK] (scratch then result)

  char* ws = (char*)d_ws;
  const size_t FEAT = (size_t)NQ * DIMD * sizeof(__bf16);   // 4 MB
  const size_t WMAT = (size_t)DIMD * DIMD * sizeof(__bf16); // 128 KB
  __bf16* qfb = (__bf16*)(ws);
  __bf16* kfb = (__bf16*)(ws + FEAT);
  __bf16* Wqb = (__bf16*)(ws + 2 * FEAT);
  __bf16* Wkb = (__bf16*)(ws + 2 * FEAT + WMAT);
  __bf16* Wvb = (__bf16*)(ws + 2 * FEAT + 2 * WMAT);
  __bf16* Wob = (__bf16*)(ws + 2 * FEAT + 3 * WMAT);
  __bf16* Qb  = (__bf16*)(ws + 2 * FEAT + 4 * WMAT);
  __bf16* Kb  = (__bf16*)(ws + 3 * FEAT + 4 * WMAT);
  __bf16* Vtb = (__bf16*)(ws + 4 * FEAT + 4 * WMAT);   // transposed [DIMD][NK]
  __bf16* Cb  = (__bf16*)(ws + 5 * FEAT + 4 * WMAT);

  const long nfeat = (long)NQ * DIMD;
  const long nw    = (long)DIMD * DIMD;
  cvt_f32_bf16<<<(unsigned)(nfeat / 8 / 256), 256, 0, stream>>>(q_feat, qfb, nfeat);
  cvt_f32_bf16<<<(unsigned)(nfeat / 8 / 256), 256, 0, stream>>>(k_feat, kfb, nfeat);
  cvt_f32_bf16<<<(unsigned)(nw / 8 / 256), 256, 0, stream>>>(Wq, Wqb, nw);
  cvt_f32_bf16<<<(unsigned)(nw / 8 / 256), 256, 0, stream>>>(Wk, Wkb, nw);
  cvt_f32_bf16<<<(unsigned)(nw / 8 / 256), 256, 0, stream>>>(Wv, Wvb, nw);
  cvt_f32_bf16<<<(unsigned)(nw / 8 / 256), 256, 0, stream>>>(Wo, Wob, nw);

  dim3 blk(256);
  dim3 gproj(DIMD / 64, NQ / 128);          // (4, 64)
  gemm_xWT_bias<<<gproj, blk, 0, stream>>>(qfb, Wqb, bq, Qb,  NQ, 0);
  gemm_xWT_bias<<<gproj, blk, 0, stream>>>(kfb, Wkb, bk, Kb,  NK, 0);
  gemm_xWT_bias<<<gproj, blk, 0, stream>>>(kfb, Wvb, bv, Vtb, NK, 1);

  dim3 gscore(NK / 128, NQ / 128);          // (64, 64)
  score_kernel<<<gscore, blk, 0, stream>>>(Qb, Kb, qbat, kbat, attn);

  softmax_kernel<<<NQ, 256, 0, stream>>>(attn);

  dim3 gctx(DIMD / 128, NQ / 128);          // (2, 64)
  ctx_kernel<<<gctx, blk, 0, stream>>>(attn, Vtb, Cb);

  gemm_xWT_bias<<<gproj, blk, 0, stream>>>(Cb, Wob, bo, attended, NQ, 2);
}
